// Attention_75565654606265
// MI455X (gfx1250) — 16-bit and 8-bit matrix operand order repaired and checked by reading, compiled, not run on hardware
//
#include <hip/hip_runtime.h>
#include <hip/hip_bf16.h>

typedef __attribute__((ext_vector_type(16))) _Float16 v16h;
typedef __attribute__((ext_vector_type(8)))  _Float16 v8h;
typedef __attribute__((ext_vector_type(8)))  float    v8f;
typedef __attribute__((ext_vector_type(16))) int      v16i;

#define T_  4
#define B_  8
#define N_  1024
#define C_  512
#define H_  8
#define D_  64
#define BN_ (B_*N_)
#define O3_ (3*C_)
#define NBC_ ((size_t)BN_*(size_t)C_)

#define FP8_ONE 0x38

#define WMMA_F16(a,b,c) __builtin_amdgcn_wmma_f32_16x16x32_f16(false,(a),false,(b),(short)0,(c),false,false)
#define WMMA_FP8(a,b,c) __builtin_amdgcn_wmma_f32_16x16x128_fp8_fp8((a),(b),(short)0,(c),false,false)

__device__ __forceinline__ float sigmoidf_(float w){ return 1.0f/(1.0f + __expf(-w)); }

__device__ __forceinline__ v16h cat16(v8h lo, v8h hi){
  return __builtin_shufflevector(lo, hi, 0,1,2,3,4,5,6,7,8,9,10,11,12,13,14,15);
}

__device__ __forceinline__ unsigned char f32_to_e4m3(float f){
  unsigned u = __float_as_uint(f);
  unsigned char sign = (unsigned char)((u >> 24) & 0x80u);
  float a = __builtin_fabsf(f);
  if (!(a >= 0.0009765625f)) return sign;
  if (a >= 464.0f)           return (unsigned char)(sign | 0x7E);
  int e8 = ((int)((u >> 23) & 0xFF) - 127) + 7;
  unsigned man = u & 0x7FFFFFu;
  if (e8 >= 1){
    unsigned m3  = man >> 20;
    unsigned rem = man & 0xFFFFFu;
    if (rem > 0x80000u || (rem == 0x80000u && (m3 & 1u))) ++m3;
    if (m3 == 8u){ m3 = 0u; ++e8; }
    if (e8 >= 16 || (e8 == 15 && m3 == 7u)) return (unsigned char)(sign | 0x7E);
    return (unsigned char)(sign | (e8 << 3) | m3);
  }
  int m3 = (int)(a * 512.0f + 0.5f);
  if (m3 > 7) return (unsigned char)(sign | 0x08);
  return (unsigned char)(sign | m3);
}

__device__ __forceinline__ v16h load_a_f16(const _Float16* src, int row0, int k0, int ld){
  int lane = threadIdx.x & 31;
  int m = lane & 15, hf = lane >> 4;
  const _Float16* p = src + (size_t)(row0 + m)*ld + k0 + hf*8;
  return cat16(*(const v8h*)p, *(const v8h*)(p + 16));
}

__device__ __forceinline__ v16i load_a_fp8(const unsigned char* src, int row0, int k0, int ld){
  int lane = threadIdx.x & 31;
  int m = lane & 15, hf = lane >> 4;
  const unsigned char* p = src + (size_t)(row0 + m)*ld + k0 + hf*8;
  v16i a;
#pragma unroll
  for (int c = 0; c < 8; ++c){
    uint2 q = *(const uint2*)(p + 16*c);
    a[2*c] = (int)q.x; a[2*c+1] = (int)q.y;
  }
  return a;
}

__device__ __forceinline__ v16i load_bt_fp8(const unsigned char* bt, int col0, int k0, int ld){
  int lane = threadIdx.x & 31;
  int j = lane & 15, hf = lane >> 4;
  const unsigned char* p = bt + (size_t)(col0 + j)*ld + k0 + hf*8;
  v16i b;
#pragma unroll
  for (int c = 0; c < 8; ++c){
    uint2 q = *(const uint2*)(p + 16*c);
    b[2*c] = (int)q.x; b[2*c+1] = (int)q.y;
  }
  return b;
}

__global__ void k_cvt_w(const float* __restrict__ wq, const float* __restrict__ wp,
                        unsigned char* __restrict__ wq8, unsigned char* __restrict__ wp8){
  int idx = blockIdx.x*256 + threadIdx.x;
  const int NQ = O3_*C_;
  if (idx < NQ) wq8[idx] = f32_to_e4m3(wq[idx]);
  int j = idx - NQ;
  if (j >= 0 && j < C_*C_) wp8[j] = f32_to_e4m3(wp[j]);
}

__global__ void k_plif_in(const float* __restrict__ x, const float* __restrict__ pw,
                          unsigned char* __restrict__ xs8){
  size_t e = (size_t)blockIdx.x*256 + threadIdx.x;
  float sg = sigmoidf_(pw[0]);
  float v = 0.f;
#pragma unroll
  for (int t = 0; t < T_; ++t){
    float xv = x[(size_t)t*NBC_ + e];
    v += (xv - v)*sg;
    int s = (v >= 1.f);
    xs8[(size_t)t*NBC_ + e] = s ? (unsigned char)FP8_ONE : (unsigned char)0;
    v = s ? 0.f : v;
  }
}

__global__ void __launch_bounds__(256) k_qkv(const unsigned char* __restrict__ xs8,
                                             const unsigned char* __restrict__ wq8,
                                             const float* __restrict__ pq,
                                             const float* __restrict__ pk,
                                             const float* __restrict__ pv,
                                             _Float16* __restrict__ qkvs){
  int wave = threadIdx.x >> 5;
  int lane = threadIdx.x & 31;
  int m = lane & 15, hf = lane >> 4;
  int r0 = blockIdx.x*128 + wave*16;
  int o0 = blockIdx.y*64;
  int seg = o0 >> 9;
  float w = (seg==0) ? pq[0] : (seg==1) ? pk[0] : pv[0];
  float sg = sigmoidf_(w);

  v8f z = {0.f,0.f,0.f,0.f,0.f,0.f,0.f,0.f};
  v8f acc[T_][4];
#pragma unroll
  for (int t=0;t<T_;++t)
#pragma unroll
    for (int j=0;j<4;++j) acc[t][j] = z;

#pragma unroll
  for (int k0 = 0; k0 < C_; k0 += 128){
    v16i bf[4];
#pragma unroll
    for (int j=0;j<4;++j) bf[j] = load_bt_fp8(wq8, o0 + j*16, k0, C_);
    __builtin_prefetch(wq8 + (size_t)o0*C_ + k0 + 256, 0, 0);
#pragma unroll
    for (int t=0;t<T_;++t){
      v16i af = load_a_fp8(xs8 + (size_t)t*NBC_, r0, k0, C_);
#pragma unroll
      for (int j=0;j<4;++j) acc[t][j] = WMMA_FP8(af, bf[j], acc[t][j]);
    }
  }

  float vst[4][8] = {};
#pragma unroll
  for (int t=0;t<T_;++t){
#pragma unroll
    for (int j=0;j<4;++j){
      int col = o0 + j*16 + m;
#pragma unroll
      for (int r=0;r<8;++r){
        float val = acc[t][j][r];
        float v = vst[j][r];
        v += (val - v)*sg;
        float s = (v >= 1.f) ? 1.f : 0.f;
        vst[j][r] = (s > 0.f) ? 0.f : v;
        int row = r0 + r + hf*8;
        qkvs[((size_t)t*BN_ + row)*O3_ + col] = (_Float16)s;
      }
    }
  }
}

__global__ void __launch_bounds__(256) k_kv(const _Float16* __restrict__ qkvs,
                                            float* __restrict__ kvbuf){
  __shared__ _Float16 KT[64*32];
  __shared__ _Float16 VT[64*32];
  int tbh = blockIdx.x;
  int t = tbh / (B_*H_);
  int b = (tbh / H_) % B_;
  int h = tbh % H_;
  size_t rowbase = (size_t)t*BN_ + (size_t)b*N_;
  int kcol = C_ + h*D_;
  int vcol = 2*C_ + h*D_;
  int tid = threadIdx.x;
  int wave = tid >> 5, lane = tid & 31, m = lane & 15, hf = lane >> 4;

  int ti = wave*2;
  int dt = (ti >> 2)*16;
  int e0 = (ti & 3)*16;
  int e1 = e0 + 16;

  v8f z = {0.f,0.f,0.f,0.f,0.f,0.f,0.f,0.f};
  v8f acc0 = z, acc1 = z;

  for (int n0 = 0; n0 < N_; n0 += 32){
    {
      int nn = tid >> 3;
      int dd = (tid & 7)*8;
      const _Float16* ps = qkvs + (rowbase + n0 + nn)*O3_;
      v8h kv8 = *(const v8h*)(ps + kcol + dd);
      v8h vv8 = *(const v8h*)(ps + vcol + dd);
#pragma unroll
      for (int i=0;i<8;++i){ KT[(dd+i)*32 + nn] = kv8[i]; VT[(dd+i)*32 + nn] = vv8[i]; }
    }
    __syncthreads();

    v8h alo = *(const v8h*)&KT[(dt + m)*32 + hf*16];
    v8h ahi = *(const v8h*)&KT[(dt + m)*32 + hf*16 + 8];
    v16h a = cat16(alo, ahi);

    v8h b0lo = *(const v8h*)&VT[(e0 + m)*32 + hf*16];
    v8h b0hi = *(const v8h*)&VT[(e0 + m)*32 + hf*16 + 8];
    acc0 = WMMA_F16(a, cat16(b0lo, b0hi), acc0);

    v8h b1lo = *(const v8h*)&VT[(e1 + m)*32 + hf*16];
    v8h b1hi = *(const v8h*)&VT[(e1 + m)*32 + hf*16 + 8];
    acc1 = WMMA_F16(a, cat16(b1lo, b1hi), acc1);

    __syncthreads();
  }

  float* out = kvbuf + (size_t)tbh*4096;
#pragma unroll
  for (int r=0;r<8;++r){
    int dr = dt + r + hf*8;
    out[dr*64 + e0 + m] = acc0[r];
    out[dr*64 + e1 + m] = acc1[r];
  }
}

__global__ void __launch_bounds__(256) k_attn_o(const _Float16* __restrict__ qkvs,
                                                const float* __restrict__ kvbuf,
                                                const float* __restrict__ pproj,
                                                unsigned char* __restrict__ os8){
  __shared__ _Float16 KVT[64*64];
  int blk = blockIdx.x;
  int b = blk / (H_*32);
  int rem = blk % (H_*32);
  int h = rem / 32;
  int n0 = (rem % 32)*32;
  int tid = threadIdx.x, wave = tid >> 5, lane = tid & 31, m = lane & 15, hf = lane >> 4;
  int ni = (wave >> 2)*16;
  int ei = (wave & 3)*16;
  float sg = sigmoidf_(pproj[0]);
  const float scale = 0.125f;
  float vst[8] = {};
  v8f z = {0.f,0.f,0.f,0.f,0.f,0.f,0.f,0.f};

#pragma unroll 1
  for (int t=0;t<T_;++t){
    int tbh = (t*B_ + b)*H_ + h;
    const float* kvsrc = kvbuf + (size_t)tbh*4096;
#pragma unroll
    for (int i=0;i<16;++i){
      int idx = tid*16 + i;
      int d = idx >> 6, e = idx & 63;
      KVT[e*64 + d] = (_Float16)kvsrc[idx];
    }
    __syncthreads();

    size_t rowbase = (size_t)t*BN_ + (size_t)b*N_ + n0 + ni;
    v8f acc = z;
#pragma unroll
    for (int kd = 0; kd < D_; kd += 32){
      const _Float16* pa = qkvs + (rowbase + m)*O3_ + h*D_ + kd + hf*8;
      v16h a = cat16(*(const v8h*)pa, *(const v8h*)(pa + 16));
      const _Float16* pb = &KVT[(ei + m)*64 + kd + hf*8];
      v16h bf = cat16(*(const v8h*)pb, *(const v8h*)(pb + 16));
      acc = WMMA_F16(a, bf, acc);
    }

    int col = h*D_ + ei + m;
#pragma unroll
    for (int r=0;r<8;++r){
      float val = acc[r]*scale;
      float v = vst[r];
      v += (val - v)*sg;
      int s = (v >= 1.f);
      vst[r] = s ? 0.f : v;
      size_t row = rowbase + r + hf*8;
      os8[row*C_ + col] = s ? (unsigned char)FP8_ONE : (unsigned char)0;
    }
    __syncthreads();
  }
}

__global__ void __launch_bounds__(256) k_proj(const unsigned char* __restrict__ os8,
                                              const unsigned char* __restrict__ wp8,
                                              const float* __restrict__ bias,
                                              float* __restrict__ out){
  int wave = threadIdx.x >> 5, lane = threadIdx.x & 31, m = lane & 15, hf = lane >> 4;
  int r0 = blockIdx.x*128 + wave*16;
  int j0 = blockIdx.y*64;
  v8f z = {0.f,0.f,0.f,0.f,0.f,0.f,0.f,0.f};
  v8f acc[4] = {z,z,z,z};
#pragma unroll
  for (int k0 = 0; k0 < C_; k0 += 128){
    v16i a = load_a_fp8(os8, r0, k0, C_);
    __builtin_prefetch(wp8 + (size_t)j0*C_ + k0 + 256, 0, 0);
#pragma unroll
    for (int j=0;j<4;++j){
      v16i bf = load_bt_fp8(wp8, j0 + j*16, k0, C_);
      acc[j] = WMMA_FP8(a, bf, acc[j]);
    }
  }
#pragma unroll
  for (int j=0;j<4;++j){
    int col = j0 + j*16 + m;
    float bb = bias[col];
#pragma unroll
    for (int r=0;r<8;++r)
      out[(size_t)(r0 + r + hf*8)*C_ + col] = acc[j][r] + bb;
  }
}

extern "C" void kernel_launch(void* const* d_in, const int* in_sizes, int n_in,
                              void* d_out, int out_size, void* d_ws, size_t ws_size,
                              hipStream_t stream) {
  (void)in_sizes; (void)n_in; (void)out_size; (void)ws_size;
  const float* x      = (const float*)d_in[0];
  const float* w_qkv  = (const float*)d_in[1];
  const float* w_proj = (const float*)d_in[2];
  const float* b_proj = (const float*)d_in[3];
  const float* p_in   = (const float*)d_in[4];
  const float* p_q    = (const float*)d_in[5];
  const float* p_k    = (const float*)d_in[6];
  const float* p_v    = (const float*)d_in[7];
  const float* p_proj = (const float*)d_in[8];
  float* out = (float*)d_out;

  char* ws = (char*)d_ws;
  size_t off = 0;
  auto take = [&](size_t bytes)->char* {
    char* p = ws + off;
    off = (off + bytes + 255) & ~(size_t)255;
    return p;
  };
  unsigned char* wq8  = (unsigned char*)take((size_t)O3_*C_);
  unsigned char* wp8  = (unsigned char*)take((size_t)C_*C_);
  unsigned char* xs8  = (unsigned char*)take((size_t)T_*NBC_);
  _Float16*      qkvs = (_Float16*)take((size_t)T_*BN_*O3_*sizeof(_Float16));
  float*         kvbf = (float*)   take((size_t)T_*B_*H_*D_*D_*sizeof(float));
  unsigned char* os8  = (unsigned char*)take((size_t)T_*NBC_);

  k_cvt_w  <<<dim3((O3_*C_ + C_*C_)/256), dim3(256), 0, stream>>>(w_qkv, w_proj, wq8, wp8);
  k_plif_in<<<dim3(NBC_/256),             dim3(256), 0, stream>>>(x, p_in, xs8);
  k_qkv    <<<dim3(BN_/128, O3_/64),      dim3(256), 0, stream>>>(xs8, wq8, p_q, p_k, p_v, qkvs);
  k_kv     <<<dim3(T_*B_*H_),             dim3(256), 0, stream>>>(qkvs, kvbf);
  k_attn_o <<<dim3(B_*H_*(N_/32)),        dim3(256), 0, stream>>>(qkvs, kvbf, p_proj, os8);
  k_proj   <<<dim3((T_*BN_)/128, C_/64),  dim3(256), 0, stream>>>(os8, wp8, b_proj, out);
}
